// STGNN_AR_GhostFusor_BG_54004918780388
// MI455X (gfx1250) — matrix operand order repaired and checked by reading, not compiled, not run on hardware
//
#include <hip/hip_runtime.h>

// ---------------------------------------------------------------------------
// STGNN autoregressive forecaster for gfx1250 (MI455X).
//  * Dense matmuls (GNN layers, GRU gates, head MLP): v_wmma_f32_16x16x32_bf16,
//    fp32 accumulate, LDS-staged A tiles shared by 4 waves/block.
//  * Edge aggregation: dst-CSR built once per launch (LDS scan), hot loop is a
//    pure register-accumulate gather with ZERO atomics.
//  * Physics / boundary MLPs: scalar VALU (latency-bound, not FLOP-bound).
// ---------------------------------------------------------------------------

typedef __attribute__((ext_vector_type(16))) __bf16 v16bf;
typedef __attribute__((ext_vector_type(8)))  __bf16 v8bf;
typedef __attribute__((ext_vector_type(8)))  float  v8f;

#define CDIV(a,b) (((a)+(b)-1)/(b))

static constexpr int Nn   = 50000;
static constexpr int Ee   = 800000;
static constexpr int Ff   = 8;
static constexpr int Hh   = 64;
static constexpr int NBn  = Nn / 10;     // 5000 boundary nodes
static constexpr int TINs = 12;
static constexpr int TOUTs= 12;
static constexpr int BGC  = 2;
static constexpr int SCB  = 512;         // scan block size

static __device__ __forceinline__ float sigm(float x){ return 1.f / (1.f + __expf(-x)); }

// ---------------------------------------------------------------------------
// Setup kernels
// ---------------------------------------------------------------------------
__global__ void k_zero(float* p, int n){
  int i = blockIdx.x * blockDim.x + threadIdx.x;
  if (i < n) p[i] = 0.f;
}

__global__ void k_ew(const float* __restrict__ ea, float* __restrict__ ew, int E){
  int e = blockIdx.x * blockDim.x + threadIdx.x;
  if (e < E) ew[e] = 1.f / fmaxf(ea[2*e], 1e-6f);
}

__global__ void k_count(const int* __restrict__ dst, const int* __restrict__ src,
                        int* cntD, int* cntS, int E){
  int e = blockIdx.x * blockDim.x + threadIdx.x;
  if (e >= E) return;
  atomicAdd(&cntD[dst[e]], 1);
  atomicAdd(&cntS[src[e]], 1);
}

// Two-level exclusive scan: per-block Hillis-Steele in LDS + serial partials.
__global__ void k_scan_block(const int* __restrict__ cnt, int* __restrict__ scanned,
                             int* __restrict__ bsum, int N){
  __shared__ int sh[SCB];
  int tid = threadIdx.x;
  int gid = blockIdx.x * SCB + tid;
  int v = (gid < N) ? cnt[gid] : 0;
  sh[tid] = v;
  __syncthreads();
  for (int off = 1; off < SCB; off <<= 1){
    int t = (tid >= off) ? sh[tid - off] : 0;
    __syncthreads();
    sh[tid] += t;
    __syncthreads();
  }
  if (gid < N) scanned[gid] = sh[tid] - v;       // exclusive
  if (tid == SCB - 1) bsum[blockIdx.x] = sh[SCB - 1];
}

__global__ void k_scan_partials(int* bsum, int nb){
  if (blockIdx.x == 0 && threadIdx.x == 0){
    int acc = 0;
    for (int i = 0; i < nb; ++i){ int v = bsum[i]; bsum[i] = acc; acc += v; }
  }
}

__global__ void k_rowptr(const int* __restrict__ scanned, const int* __restrict__ bsum,
                         int* __restrict__ rowptr, int N, int E){
  int i = blockIdx.x * blockDim.x + threadIdx.x;
  if (i < N) rowptr[i] = scanned[i] + bsum[i / SCB];
  if (i == 0) rowptr[N] = E;
}

__global__ void k_copy_int(const int* __restrict__ a, int* __restrict__ b, int n){
  int i = blockIdx.x * blockDim.x + threadIdx.x;
  if (i < n) b[i] = a[i];
}

__global__ void k_fill(const int* __restrict__ dst, int* cursor, int* __restrict__ eid, int E){
  int e = blockIdx.x * blockDim.x + threadIdx.x;
  if (e >= E) return;
  int p = atomicAdd(&cursor[dst[e]], 1);
  eid[p] = e;
}

// Flatten per-slot (source node, edge weight) so the hot gather has 1 indirection.
__global__ void k_csr_prep(const int* __restrict__ eid, const int* __restrict__ src,
                           const float* __restrict__ ew, int* __restrict__ csrc,
                           float* __restrict__ cw, int E){
  int j = blockIdx.x * blockDim.x + threadIdx.x;
  if (j >= E) return;
  int e = eid[j];
  csrc[j] = src[e];
  cw[j]   = ew[e];
}

__global__ void k_node_pre(const int* __restrict__ rowptr, const float* __restrict__ cw,
                           const int* __restrict__ cntS, float* __restrict__ invdegw,
                           float* __restrict__ dt_deg, const float* __restrict__ dtp, int N){
  int n = blockIdx.x * blockDim.x + threadIdx.x;
  if (n >= N) return;
  int j0 = rowptr[n], j1 = rowptr[n + 1];
  float dw = 0.f;
  for (int j = j0; j < j1; ++j) dw += cw[j];
  invdegw[n] = 1.f / fmaxf(dw, 1e-6f);
  dt_deg[n]  = dtp[0] / fmaxf((float)((j1 - j0) + cntS[n]), 1.f);
}

__global__ void k_bnd_pre(const float* __restrict__ ea, const int* __restrict__ bfe,
                          float* inv_dxb, int nb){
  int i = blockIdx.x * blockDim.x + threadIdx.x;
  if (i >= nb) return;
  inv_dxb[i] = 1.f / fmaxf(ea[2*bfe[i]], 1e-6f);
}

// Pack weight W[K][Nout] (f32, row-major) -> WT[Nout][Kp] (bf16, zero-padded K).
__global__ void k_pack_wt(const float* __restrict__ W, __bf16* __restrict__ WT,
                          int K, int Nout, int Kp){
  int i = blockIdx.x * blockDim.x + threadIdx.x;
  if (i >= Nout * Kp) return;
  int n = i / Kp, k = i % Kp;
  WT[i] = (k < K) ? (__bf16)W[k * Nout + n] : (__bf16)0.f;
}

// ---------------------------------------------------------------------------
// CSR gather (no atomics): agg[n][f] = sum_{incoming j} cw[j] * h[csrc[j]][f]
// ---------------------------------------------------------------------------
__global__ void k_gather_csr(const float* __restrict__ hsrc, const int* __restrict__ rowptr,
                             const int* __restrict__ csrc, const float* __restrict__ cw,
                             float* __restrict__ agg, int N, int width){
  int i = blockIdx.x * blockDim.x + threadIdx.x;
  if (i >= N * width) return;
  int n = i / width, f = i % width;
  int j0 = rowptr[n], j1 = rowptr[n + 1];
  float acc = 0.f;
  for (int j = j0; j < j1; ++j)
    acc += cw[j] * hsrc[(size_t)csrc[j] * width + f];
  agg[i] = acc;
}

// ---------------------------------------------------------------------------
// bf16 A-matrix staging kernels (pad K -> Kp)
// ---------------------------------------------------------------------------
__global__ void k_stage_combine(const float* __restrict__ hv, const float* __restrict__ agg,
                                const float* __restrict__ invdegw, __bf16* __restrict__ out,
                                int K, int Kp, int N){
  int i = blockIdx.x * blockDim.x + threadIdx.x;
  if (i >= N * Kp) return;
  int n = i / Kp, k = i % Kp;
  float v = (k < K) ? hv[(size_t)n * K + k] + agg[(size_t)n * K + k] * invdegw[n] : 0.f;
  out[i] = (__bf16)v;
}

__global__ void k_stage_pad(const float* __restrict__ in, __bf16* __restrict__ out,
                            int K, int Kp, int N){
  int i = blockIdx.x * blockDim.x + threadIdx.x;
  if (i >= N * Kp) return;
  int n = i / Kp, k = i % Kp;
  out[i] = (__bf16)((k < K) ? in[(size_t)n * K + k] : 0.f);
}

__global__ void k_bcast8(const float* __restrict__ cur, float* __restrict__ g, int N){
  int i = blockIdx.x * blockDim.x + threadIdx.x;
  if (i < N * Ff) g[i] = cur[i >> 3];
}

// ---------------------------------------------------------------------------
// WMMA GEMM:  out[N x Nout] = act( A[N x Kp](bf16) @ WT^T + bias )
//  - A tile (16 rows x Kp, contiguous 2KB) staged into LDS once per block,
//    shared by the 4 waves (4 n-tiles) -> fragments come from ds_load_b128.
//  - WT stored [Nout][Kp] bf16 so B fragments are contiguous 32B global loads.
// ---------------------------------------------------------------------------
__global__ void __launch_bounds__(128)
k_wmma_gemm(const __bf16* __restrict__ A, const __bf16* __restrict__ WT,
            const float* __restrict__ bias, float* __restrict__ out,
            int Kp, int Nout, int relu){
  __shared__ __align__(16) __bf16 As[16 * 64];   // max Kp = 64 -> 2KB
  const int lane = threadIdx.x;
  const int m0   = blockIdx.x * 16;

  // Cooperative stage: 16 x Kp bf16 = contiguous block starting at A + m0*Kp.
  {
    const unsigned int* ag = (const unsigned int*)(A + (size_t)m0 * Kp);
    unsigned int* as = (unsigned int*)As;
    int tid  = threadIdx.y * 32 + lane;
    int nu32 = (16 * Kp) >> 1;
    for (int idx = tid; idx < nu32; idx += 128) as[idx] = ag[idx];
  }
  __syncthreads();

  const int ntile = blockIdx.y * blockDim.y + threadIdx.y;   // wave-uniform
  if (ntile * 16 < Nout){                                    // uniform: EXEC stays all-ones
    const int ph = lane >> 4;      // half of the wave
    const int lm = lane & 15;

    // C/D layout: lane holds column n = lm; vgpr v holds row m = v + 8*ph.
    const float bn = bias[ntile * 16 + lm];
    v8f acc;
#pragma unroll
    for (int v = 0; v < 8; ++v) acc[v] = bn;

    const __bf16* arow = As + lm * Kp;                                 // A: lane = row M
    const __bf16* brow = WT + (size_t)(ntile * 16 + lm) * Kp + ph * 16;// B: lane = col N

    for (int k0 = 0; k0 < Kp; k0 += 32){
      // A fragment: elems 0..15 = K {k0+ph*16 ..} (contiguous, matches B)
      v16bf a = *(const v16bf*)(arow + k0 + ph * 16);
      // B fragment: 16 contiguous K values for this lane's output column
      v16bf b = *(const v16bf*)(brow + k0);
      acc = __builtin_amdgcn_wmma_f32_16x16x32_bf16(false, a, false, b, (short)0,
                                                    acc, false, false);
    }

    float* orow = out + (size_t)(m0 + 8 * ph) * Nout + ntile * 16 + lm;
#pragma unroll
    for (int v = 0; v < 8; ++v){
      float val = acc[v];
      if (relu) val = fmaxf(val, 0.f);
      orow[(size_t)v * Nout] = val;
    }
  }
}

// ---------------------------------------------------------------------------
// GRU gate math (gi/gh already contain biases from the GEMMs)
// ---------------------------------------------------------------------------
__global__ void k_gru(const float* __restrict__ gi, const float* __restrict__ gh,
                      const float* __restrict__ h, float* __restrict__ hn, int N){
  int i = blockIdx.x * blockDim.x + threadIdx.x;
  if (i >= N * Hh) return;
  int n = i >> 6, j = i & 63;
  const float* gip = gi + (size_t)n * 3 * Hh;
  const float* ghp = gh + (size_t)n * 3 * Hh;
  float r  = sigm(gip[j]        + ghp[j]);
  float z  = sigm(gip[Hh + j]   + ghp[Hh + j]);
  float nn = tanhf(gip[2*Hh + j] + r * ghp[2*Hh + j]);
  hn[i] = (1.f - z) * nn + z * h[i];
}

// ---------------------------------------------------------------------------
// Encoder input slice + boundary feature fuser (3F=24 -> 16 -> 8 MLP)
// ---------------------------------------------------------------------------
__global__ void k_copy_xt(const float* __restrict__ x, float* __restrict__ xf, int t, int N){
  int i = blockIdx.x * blockDim.x + threadIdx.x;
  if (i >= N * Ff) return;
  int n = i >> 3, f = i & 7;
  xf[i] = x[((size_t)n * TINs + t) * Ff + f];
}

__global__ void k_fuse_enc(float* __restrict__ xf, const int* __restrict__ bnd,
                           const int* __restrict__ down, const float* __restrict__ inv_dxb,
                           const float* __restrict__ W0, const float* __restrict__ b0,
                           const float* __restrict__ W1, const float* __restrict__ b1, int nb){
  int i = blockIdx.x * blockDim.x + threadIdx.x;
  if (i >= nb) return;
  int b = bnd[i], d = down[i];
  float sf[8], z0[24];
  float ivx = inv_dxb[i];
#pragma unroll
  for (int f = 0; f < 8; ++f){
    sf[f] = xf[(size_t)b * 8 + f];
    float df = xf[(size_t)d * 8 + f];
    z0[f] = sf[f]; z0[8 + f] = df; z0[16 + f] = (sf[f] - df) * ivx;
  }
  float hid[16];
#pragma unroll
  for (int j = 0; j < 16; ++j){
    float a = b0[j];
    for (int k = 0; k < 24; ++k) a += z0[k] * W0[k * 16 + j];
    hid[j] = fmaxf(a, 0.f);
  }
#pragma unroll
  for (int f = 0; f < 8; ++f){
    float a = b1[f];
    for (int j = 0; j < 16; ++j) a += hid[j] * W1[j * 8 + f];
    xf[(size_t)b * 8 + f] = sf[f] + 0.5f * a;   // ALPHA_B = 0.5
  }
}

__global__ void k_extract_cur(const float* __restrict__ xf, float* __restrict__ cur, int N){
  int n = blockIdx.x * blockDim.x + threadIdx.x;
  if (n < N) cur[n] = xf[(size_t)n * Ff];
}

// ---------------------------------------------------------------------------
// Head output: y = t1 @ W2 + b2 ; also scatter into d_out (N x TOUT)
// ---------------------------------------------------------------------------
__global__ void k_head_out(const float* __restrict__ a, const float* __restrict__ w,
                           const float* __restrict__ b, float* __restrict__ y,
                           float* __restrict__ out, int t, int N){
  int n = blockIdx.x * blockDim.x + threadIdx.x;
  if (n >= N) return;
  float s = b[0];
  const float* row = a + (size_t)n * Hh;
#pragma unroll 8
  for (int k = 0; k < Hh; ++k) s += row[k] * w[k];
  y[n] = s;
  out[(size_t)n * TOUTs + t] = s;
}

__global__ void k_copy(const float* __restrict__ a, float* __restrict__ b, int n){
  int i = blockIdx.x * blockDim.x + threadIdx.x;
  if (i < n) b[i] = a[i];
}

// Decoder boundary fuser (3 -> 32 -> 1 MLP), scalar field
__global__ void k_fuse_dec(const float* __restrict__ y, float* __restrict__ nxt,
                           const int* __restrict__ bnd, const int* __restrict__ down,
                           const float* __restrict__ inv_dxb,
                           const float* __restrict__ W0, const float* __restrict__ b0,
                           const float* __restrict__ W1, const float* __restrict__ b1, int nb){
  int i = blockIdx.x * blockDim.x + threadIdx.x;
  if (i >= nb) return;
  int b = bnd[i], d = down[i];
  float sf = y[b], df = y[d];
  float diff = (sf - df) * inv_dxb[i];
  float delta = b1[0];
#pragma unroll 8
  for (int j = 0; j < 32; ++j){
    float a = b0[j] + sf * W0[j] + df * W0[32 + j] + diff * W0[64 + j];
    delta += fmaxf(a, 0.f) * W1[j];
  }
  nxt[b] = sf + 0.5f * delta;   // ALPHA_B
}

__global__ void k_anchor(const float* __restrict__ nxt, const int* __restrict__ bnd,
                         float* __restrict__ anc, int nb){
  int i = blockIdx.x * blockDim.x + threadIdx.x;
  if (i < nb) anc[i] = nxt[bnd[i]];
}

__global__ void k_relax(float* __restrict__ nxt, const int* __restrict__ bnd,
                        const float* __restrict__ anc, int nb){
  int i = blockIdx.x * blockDim.x + threadIdx.x;
  if (i >= nb) return;
  int b = bnd[i];
  nxt[b] = 0.3f * nxt[b] + 0.7f * anc[i];       // (1-RELAX), RELAX
}

// Physics step: flux over edges, +/- scatter, explicit update
__global__ void k_phys_edge(const float* __restrict__ v, const int* __restrict__ src,
                            const int* __restrict__ dst, const float* __restrict__ ea,
                            const float* __restrict__ ew, float* __restrict__ aggv, int E){
  int e = blockIdx.x * blockDim.x + threadIdx.x;
  if (e >= E) return;
  int s = src[e], d = dst[e];
  float f = (v[s] - v[d] + ea[2*e + 1]) * ew[e];
  atomicAdd(&aggv[d],  f);
  atomicAdd(&aggv[s], -f);
}

__global__ void k_phys_node(float* __restrict__ v, const float* __restrict__ aggv,
                            const float* __restrict__ dt_deg, int N){
  int n = blockIdx.x * blockDim.x + threadIdx.x;
  if (n < N) v[n] += dt_deg[n] * aggv[n];
}

// ---------------------------------------------------------------------------
// Host orchestration
// ---------------------------------------------------------------------------
extern "C" void kernel_launch(void* const* d_in, const int* in_sizes, int n_in,
                              void* d_out, int out_size, void* d_ws, size_t ws_size,
                              hipStream_t stream) {
  (void)in_sizes; (void)n_in; (void)out_size; (void)ws_size;
  // ---- inputs (setup_inputs flatten order) ----
  const float* x      = (const float*)d_in[0];
  const float* ea     = (const float*)d_in[1];
  const float* gW0    = (const float*)d_in[2];  const float* gb0 = (const float*)d_in[3];
  const float* gW1    = (const float*)d_in[4];  const float* gb1 = (const float*)d_in[5];
  const float* gW2    = (const float*)d_in[6];  const float* gb2 = (const float*)d_in[7];
  const float* Wih    = (const float*)d_in[8];  const float* Whh = (const float*)d_in[9];
  const float* bih    = (const float*)d_in[10]; const float* bhh = (const float*)d_in[11];
  const float* hW0    = (const float*)d_in[12]; const float* hb0 = (const float*)d_in[13];
  const float* hW1    = (const float*)d_in[14]; const float* hb1 = (const float*)d_in[15];
  const float* hW2    = (const float*)d_in[16]; const float* hb2 = (const float*)d_in[17];
  const float* feW0   = (const float*)d_in[18]; const float* feb0= (const float*)d_in[19];
  const float* feW1   = (const float*)d_in[20]; const float* feb1= (const float*)d_in[21];
  const float* fdW0   = (const float*)d_in[22]; const float* fdb0= (const float*)d_in[23];
  const float* fdW1   = (const float*)d_in[24]; const float* fdb1= (const float*)d_in[25];
  const float* dtp    = (const float*)d_in[26];
  const int*   src    = (const int*)d_in[27];
  const int*   dst    = (const int*)d_in[28];
  const int*   bnd    = (const int*)d_in[29];
  const int*   down   = (const int*)d_in[30];
  const int*   bfe    = (const int*)d_in[31];
  float* out = (float*)d_out;

  // ---- workspace carve-up (all chunks 32B-aligned multiples) ----
  size_t o = 0;
  auto alloc = [&](size_t nf) -> float* { float* p = (float*)d_ws + o; o += nf; return p; };
  float* ew      = alloc(Ee);
  float* cw      = alloc(Ee);                   // CSR-ordered edge weights
  int*   csrc    = (int*)alloc(Ee);             // CSR-ordered source nodes
  int*   csr_eid = (int*)alloc(Ee);
  int*   rowptr  = (int*)alloc(Nn + 8);
  int*   cursor  = (int*)alloc(Nn);
  int*   cntDi   = (int*)alloc(Nn);
  int*   cntSi   = (int*)alloc(Nn);
  int*   scanned = (int*)alloc(Nn);
  int*   bsum    = (int*)alloc(128);
  float* invdegw = alloc(Nn);
  float* dt_deg  = alloc(Nn);
  float* inv_dxb = alloc(NBn);
  float* h       = alloc((size_t)Nn * Hh);
  float* hb      = alloc((size_t)Nn * Hh);
  float* act0    = alloc((size_t)Nn * Hh);
  float* act1    = alloc((size_t)Nn * Hh);
  float* agg     = alloc((size_t)Nn * Hh);
  float* gi      = alloc((size_t)Nn * 3 * Hh);
  float* gh      = alloc((size_t)Nn * 3 * Hh);
  float* xf      = alloc((size_t)Nn * Ff);
  float* gbuf    = alloc((size_t)Nn * Ff);
  float* ybuf    = alloc(Nn);
  float* curA    = alloc(Nn);
  float* curB    = alloc(Nn);
  float* anchor  = alloc(NBn);
  float* aggv    = alloc(Nn);
  __bf16* stage  = (__bf16*)alloc((size_t)Nn * Hh / 2);   // N x 64 bf16
  __bf16* WT0    = (__bf16*)alloc(64 * 32 / 2);           // [64][32]
  __bf16* WT1    = (__bf16*)alloc(64 * 64 / 2);
  __bf16* WT2    = (__bf16*)alloc(64 * 64 / 2);
  __bf16* WTih   = (__bf16*)alloc(192 * 64 / 2);
  __bf16* WThh   = (__bf16*)alloc(192 * 64 / 2);
  __bf16* WTh0   = (__bf16*)alloc(64 * 64 / 2);
  __bf16* WTh1   = (__bf16*)alloc(64 * 64 / 2);

  const int BS = 256;
  auto zero = [&](float* p, int n){ k_zero<<<CDIV(n,BS), BS, 0, stream>>>(p, n); };

  // ---- weight packing (once per launch; deterministic) ----
  k_pack_wt<<<CDIV(64*32,BS),  BS, 0, stream>>>(gW0, WT0,  Ff, 64, 32);
  k_pack_wt<<<CDIV(64*64,BS),  BS, 0, stream>>>(gW1, WT1,  Hh, 64, 64);
  k_pack_wt<<<CDIV(64*64,BS),  BS, 0, stream>>>(gW2, WT2,  Hh, 64, 64);
  k_pack_wt<<<CDIV(192*64,BS), BS, 0, stream>>>(Wih, WTih, Hh, 192, 64);
  k_pack_wt<<<CDIV(192*64,BS), BS, 0, stream>>>(Whh, WThh, Hh, 192, 64);
  k_pack_wt<<<CDIV(64*64,BS),  BS, 0, stream>>>(hW0, WTh0, Hh, 64, 64);
  k_pack_wt<<<CDIV(64*64,BS),  BS, 0, stream>>>(hW1, WTh1, Hh, 64, 64);

  // ---- graph precompute: edge weights + dst-CSR (atomic-free hot path) ----
  const int nScanBlocks = CDIV(Nn, SCB);
  k_ew<<<CDIV(Ee,BS), BS, 0, stream>>>(ea, ew, Ee);
  zero((float*)cntDi, Nn);
  zero((float*)cntSi, Nn);
  k_count<<<CDIV(Ee,BS), BS, 0, stream>>>(dst, src, cntDi, cntSi, Ee);
  k_scan_block<<<nScanBlocks, SCB, 0, stream>>>(cntDi, scanned, bsum, Nn);
  k_scan_partials<<<1, 32, 0, stream>>>(bsum, nScanBlocks);
  k_rowptr<<<CDIV(Nn+1,BS), BS, 0, stream>>>(scanned, bsum, rowptr, Nn, Ee);
  k_copy_int<<<CDIV(Nn,BS), BS, 0, stream>>>(rowptr, cursor, Nn);
  k_fill<<<CDIV(Ee,BS), BS, 0, stream>>>(dst, cursor, csr_eid, Ee);
  k_csr_prep<<<CDIV(Ee,BS), BS, 0, stream>>>(csr_eid, src, ew, csrc, cw, Ee);
  k_node_pre<<<CDIV(Nn,BS), BS, 0, stream>>>(rowptr, cw, cntSi, invdegw, dt_deg, dtp, Nn);
  k_bnd_pre<<<CDIV(NBn,BS), BS, 0, stream>>>(ea, bfe, inv_dxb, NBn);

  auto gemm = [&](const __bf16* A, const __bf16* WT, const float* bias, float* outp,
                  int Kp, int Nout, int relu){
    dim3 g(Nn / 16, CDIV(Nout, 64));
    dim3 b(32, 4);
    k_wmma_gemm<<<g, b, 0, stream>>>(A, WT, bias, outp, Kp, Nout, relu);
  };

  auto gnn_layer = [&](const float* hin, int K, const __bf16* WT, const float* bias,
                       float* hout){
    int Kp = (K <= 8) ? 32 : 64;
    k_gather_csr<<<CDIV(Nn*K,BS), BS, 0, stream>>>(hin, rowptr, csrc, cw, agg, Nn, K);
    k_stage_combine<<<CDIV(Nn*Kp,BS), BS, 0, stream>>>(hin, agg, invdegw, stage, K, Kp, Nn);
    gemm(stage, WT, bias, hout, Kp, 64, 1);
  };

  auto gru_step = [&](const float* g){          // swaps h <-> hb
    k_stage_pad<<<CDIV(Nn*64,BS), BS, 0, stream>>>(g, stage, Hh, 64, Nn);
    gemm(stage, WTih, bih, gi, 64, 192, 0);
    k_stage_pad<<<CDIV(Nn*64,BS), BS, 0, stream>>>(h, stage, Hh, 64, Nn);
    gemm(stage, WThh, bhh, gh, 64, 192, 0);
    k_gru<<<CDIV(Nn*Hh,BS), BS, 0, stream>>>(gi, gh, h, hb, Nn);
    float* t = h; h = hb; hb = t;
  };

  // ---- encoder ----
  zero(h, Nn * Hh);
  for (int t = 0; t < TINs; ++t){
    k_copy_xt<<<CDIV(Nn*Ff,BS), BS, 0, stream>>>(x, xf, t, Nn);
    k_fuse_enc<<<CDIV(NBn,BS), BS, 0, stream>>>(xf, bnd, down, inv_dxb,
                                                feW0, feb0, feW1, feb1, NBn);
    gnn_layer(xf,   Ff, WT0, gb0, act0);
    gnn_layer(act0, Hh, WT1, gb1, act1);
    gnn_layer(act1, Hh, WT2, gb2, act0);
    gru_step(act0);
  }
  k_extract_cur<<<CDIV(Nn,BS), BS, 0, stream>>>(xf, curA, Nn);

  // ---- decoder ----
  float* cur = curA;
  float* nxt = curB;
  for (int t = 0; t < TOUTs; ++t){
    k_bcast8<<<CDIV(Nn*Ff,BS), BS, 0, stream>>>(cur, gbuf, Nn);
    gnn_layer(gbuf, Ff, WT0, gb0, act0);
    gnn_layer(act0, Hh, WT1, gb1, act1);
    gnn_layer(act1, Hh, WT2, gb2, act0);
    gru_step(act0);
    // head MLP
    k_stage_pad<<<CDIV(Nn*64,BS), BS, 0, stream>>>(h, stage, Hh, 64, Nn);
    gemm(stage, WTh0, hb0, act0, 64, 64, 1);
    k_stage_pad<<<CDIV(Nn*64,BS), BS, 0, stream>>>(act0, stage, Hh, 64, Nn);
    gemm(stage, WTh1, hb1, act1, 64, 64, 1);
    k_head_out<<<CDIV(Nn,BS), BS, 0, stream>>>(act1, hW2, hb2, ybuf, out, t, Nn);
    // boundary fuse -> nxt
    k_copy<<<CDIV(Nn,BS), BS, 0, stream>>>(ybuf, nxt, Nn);
    k_fuse_dec<<<CDIV(NBn,BS), BS, 0, stream>>>(ybuf, nxt, bnd, down, inv_dxb,
                                                fdW0, fdb0, fdW1, fdb1, NBn);
    k_anchor<<<CDIV(NBn,BS), BS, 0, stream>>>(nxt, bnd, anchor, NBn);
    // background relaxation cycles + physics
    for (int c = 0; c < BGC; ++c){
      k_relax<<<CDIV(NBn,BS), BS, 0, stream>>>(nxt, bnd, anchor, NBn);
      zero(aggv, Nn);
      k_phys_edge<<<CDIV(Ee,BS), BS, 0, stream>>>(nxt, src, dst, ea, ew, aggv, Ee);
      k_phys_node<<<CDIV(Nn,BS), BS, 0, stream>>>(nxt, aggv, dt_deg, Nn);
    }
    float* tmp = cur; cur = nxt; nxt = tmp;
  }
}
